// CausalSelfAttention_79396765434593
// MI455X (gfx1250) — matrix operand order repaired and checked by reading, not compiled, not run on hardware
//
#include <hip/hip_runtime.h>
#include <hip/hip_bf16.h>

typedef __attribute__((ext_vector_type(16))) _Float16 v16h;
typedef __attribute__((ext_vector_type(8)))  _Float16 h8;
typedef __attribute__((ext_vector_type(4)))  _Float16 h4;
typedef __attribute__((ext_vector_type(8)))  float    v8f;

// ---------------------------------------------------------------------------
// CDNA5 async copy: global -> LDS, 16 bytes per lane, tracked by ASYNCcnt.
// lds_off is the byte offset within LDS (low 32 bits of a generic LDS pointer).
// ---------------------------------------------------------------------------
__device__ __forceinline__ void async_copy_b128(const void* gaddr, unsigned lds_off) {
    asm volatile("global_load_async_to_lds_b128 %0, %1, off"
                 :: "v"(lds_off), "v"(gaddr) : "memory");
}
__device__ __forceinline__ void async_wait0() {
    asm volatile("s_wait_asynccnt 0x0" ::: "memory");
}
__device__ __forceinline__ unsigned lds_off_of(const void* p) {
    return (unsigned)(size_t)p;   // generic LDS pointer: low 32 bits = LDS offset
}

// ---------------------------------------------------------------------------
// Fragment loaders following CDNA5 WMMA 16-bit layouts (CDNA5 ISA).
// A 16x32 f16: lanes 0-15 -> row=lane, K = {kb..kb+7, kb+16..kb+23}, kb = 0|8.
// B 32x16 f16: lanes 0-15 -> col=lane, K = {kb..kb+7, kb+16..kb+23}, kb = 0|8.
// C/D 16x16 f32: VGPR r -> row = 8*(lane>=16)+r, col = lane&15.
// ---------------------------------------------------------------------------
__device__ __forceinline__ v16h frag_a(const _Float16* tile, int lane, int stride) {
    const int row = lane & 15;
    const int kb  = (lane >> 4) << 3;
    const _Float16* p = tile + row * stride + kb;
    h8 lo = *(const h8*)p;
    h8 hi = *(const h8*)(p + 16);
    v16h f;
#pragma unroll
    for (int i = 0; i < 8; ++i) { f[i] = lo[i]; f[i + 8] = hi[i]; }
    return f;
}

__device__ __forceinline__ v16h frag_a_scaled(const _Float16* tile, int lane, int stride, float s) {
    v16h f = frag_a(tile, lane, stride);
#pragma unroll
    for (int i = 0; i < 16; ++i) f[i] = (_Float16)((float)f[i] * s);
    return f;
}

__device__ __forceinline__ v16h frag_b(const _Float16* tile, int lane, int stride) {
    const int col = lane & 15;
    const int kb  = (lane >> 4) << 3;
    const _Float16* p = tile + col * stride + kb;
    h8 lo = *(const h8*)p;
    h8 hi = *(const h8*)(p + 16);
    v16h f;
#pragma unroll
    for (int i = 0; i < 8; ++i) { f[i] = lo[i]; f[i + 8] = hi[i]; }
    return f;
}

// ---------------------------------------------------------------------------
// fp32 -> fp16 conversion (one-time, bandwidth-trivial)
// ---------------------------------------------------------------------------
__global__ __launch_bounds__(256) void cvt_f32_to_f16(const float* __restrict__ in,
                                                      _Float16* __restrict__ out, int n4) {
    int i = blockIdx.x * 256 + threadIdx.x;
    if (i < n4) {
        float4 v = ((const float4*)in)[i];
        h4 o;
        o[0] = (_Float16)v.x; o[1] = (_Float16)v.y;
        o[2] = (_Float16)v.z; o[3] = (_Float16)v.w;
        ((h4*)out)[i] = o;
    }
}

// ---------------------------------------------------------------------------
// C[M,N] = A[M,K] @ W[N,K]^T + bias  (torch-Linear layout), f16 WMMA, f32 acc.
// 128x128 block tile, 8 waves, each wave 32(M) x 64(N), K-step 64.
// Tiles filled via global_load_async_to_lds_b128.
// ---------------------------------------------------------------------------
template <typename OutT>
__global__ __launch_bounds__(256) void
gemm_bias_wmma(const _Float16* __restrict__ A,   // [M,K] f16
               const _Float16* __restrict__ W,   // [N,K] f16
               const float*    __restrict__ bias,// [N]
               OutT*           __restrict__ Cout,// [M,N]
               int M, int N, int K) {
    constexpr int LDT = 72;                      // 64 + 8 pad (halves)
    __shared__ _Float16 As[128 * LDT];
    __shared__ _Float16 Ws[128 * LDT];

    const int tid  = threadIdx.x;
    const int lane = tid & 31;
    const int wave = tid >> 5;
    const int wm   = (wave & 3) * 32;
    const int wn   = (wave >> 2) * 64;
    const int bm   = blockIdx.y * 128;
    const int bn   = blockIdx.x * 128;

    v8f acc[2][4];
#pragma unroll
    for (int i = 0; i < 2; ++i)
#pragma unroll
        for (int j = 0; j < 4; ++j)
#pragma unroll
            for (int r = 0; r < 8; ++r) acc[i][j][r] = 0.0f;

    for (int kt = 0; kt < K; kt += 64) {
        __syncthreads();
#pragma unroll
        for (int i = 0; i < 4; ++i) {
            int c   = tid + i * 256;             // 1024 chunks of 8 halves (128 x 8)
            int row = c >> 3;
            int c8  = (c & 7) << 3;
            async_copy_b128(&A[(size_t)(bm + row) * K + kt + c8],
                            lds_off_of(&As[row * LDT + c8]));
            async_copy_b128(&W[(size_t)(bn + row) * K + kt + c8],
                            lds_off_of(&Ws[row * LDT + c8]));
        }
        async_wait0();
        __syncthreads();

#pragma unroll
        for (int kk = 0; kk < 64; kk += 32) {
            v16h af[2], bf[4];
#pragma unroll
            for (int i = 0; i < 2; ++i) af[i] = frag_a(&As[(wm + i * 16) * LDT + kk], lane, LDT);
#pragma unroll
            for (int j = 0; j < 4; ++j) bf[j] = frag_b(&Ws[(wn + j * 16) * LDT + kk], lane, LDT);
#pragma unroll
            for (int i = 0; i < 2; ++i)
#pragma unroll
                for (int j = 0; j < 4; ++j)
                    acc[i][j] = __builtin_amdgcn_wmma_f32_16x16x32_f16(
                        false, af[i], false, bf[j], (short)0, acc[i][j], false, false);
        }
    }

    const int mloc = (lane >> 4) << 3;
    const int nloc = lane & 15;
#pragma unroll
    for (int j = 0; j < 4; ++j) {
        int n = bn + wn + j * 16 + nloc;
        float bv = bias[n];
#pragma unroll
        for (int i = 0; i < 2; ++i)
#pragma unroll
            for (int r = 0; r < 8; ++r) {
                int m = bm + wm + i * 16 + mloc + r;
                Cout[(size_t)m * N + n] = (OutT)(acc[i][j][r] + bv);
            }
    }
}

// ---------------------------------------------------------------------------
// Flash attention: per (b, h, 128-query tile). 8 waves x 16 query rows.
// Key/value tiles of 64; causal tiles beyond query tile skipped.
// qkv layout: [B*T, 3C] halves with Q at +0, K at +C, V at +2C, head at h*D.
// K tile filled via async copy; V transposed into LDS for contiguous B-frags.
// ---------------------------------------------------------------------------
__global__ __launch_bounds__(256) void
flash_attn_wmma(const _Float16* __restrict__ qkv,  // [B*T, 3072] f16
                _Float16*       __restrict__ Y) {  // [B*T, 1024] f16
    constexpr int T = 2048, D = 64, C3 = 3072, C = 1024;
    constexpr int KS = 72;                         // K/Vt LDS stride (halves)
    constexpr int PS = 40;                         // P LDS stride (halves)
    __shared__ _Float16 Ks[64 * KS];
    __shared__ _Float16 Vt[64 * KS];
    __shared__ _Float16 Ps[8][16 * PS];

    const int tid  = threadIdx.x;
    const int lane = tid & 31;
    const int wave = tid >> 5;
    const int qt   = blockIdx.x & 15;              // T/128 = 16
    const int bh   = blockIdx.x >> 4;
    const int b    = bh >> 4;                      // H = 16
    const int h    = bh & 15;

    const int qRow0 = qt * 128 + wave * 16;
    const _Float16* qbase = qkv + (size_t)(b * T + qRow0) * C3 + h * D;

    v16h qf0 = frag_a_scaled(qbase,      lane, C3, 0.125f);  // 1/sqrt(64) folded in
    v16h qf1 = frag_a_scaled(qbase + 32, lane, C3, 0.125f);

    float mrow[8], lrow[8];
    v8f oacc[4];
#pragma unroll
    for (int r = 0; r < 8; ++r) { mrow[r] = -1e30f; lrow[r] = 0.0f; }
#pragma unroll
    for (int j = 0; j < 4; ++j)
#pragma unroll
        for (int r = 0; r < 8; ++r) oacc[j][r] = 0.0f;

    const int mloc = (lane >> 4) << 3;
    const int nloc = lane & 15;
    const int kTiles = (qt + 1) * 2;               // causal bound

    for (int kt = 0; kt < kTiles; ++kt) {
        const int k0 = kt * 64;
        __syncthreads();
        // K tile: async straight to LDS. V tile: load + transposed store.
#pragma unroll
        for (int i = 0; i < 2; ++i) {
            int c   = tid + i * 256;               // 512 chunks of 8 halves
            int row = c >> 3;
            int c8  = (c & 7) << 3;
            const size_t rbase = (size_t)(b * T + k0 + row) * C3 + h * D + c8;
            async_copy_b128(&qkv[rbase + C], lds_off_of(&Ks[row * KS + c8]));
            h8 v = *(const h8*)&qkv[rbase + 2 * C];
#pragma unroll
            for (int j = 0; j < 8; ++j) Vt[(c8 + j) * KS + row] = v[j];
        }
        async_wait0();
        __syncthreads();

        // S = (Q/sqrt(D)) K^T : 16 x 64 in 4 n-subtiles, K-dim = D = 2x32
        v8f s[4];
#pragma unroll
        for (int ns = 0; ns < 4; ++ns) {
            v16h b0 = frag_b(&Ks[(ns * 16) * KS],      lane, KS);
            v16h b1 = frag_b(&Ks[(ns * 16) * KS + 32], lane, KS);
            v8f z;
#pragma unroll
            for (int r = 0; r < 8; ++r) z[r] = 0.0f;
            z = __builtin_amdgcn_wmma_f32_16x16x32_f16(false, qf0, false, b0, (short)0, z, false, false);
            z = __builtin_amdgcn_wmma_f32_16x16x32_f16(false, qf1, false, b1, (short)0, z, false, false);
            s[ns] = z;
        }

        // causal mask + online softmax per row slot
#pragma unroll
        for (int r = 0; r < 8; ++r) {
            const int q = qRow0 + mloc + r;
            float rmax = -1e30f;
#pragma unroll
            for (int ns = 0; ns < 4; ++ns) {
                int key = k0 + ns * 16 + nloc;
                float v = (key <= q) ? s[ns][r] : -1e30f;
                s[ns][r] = v;
                rmax = fmaxf(rmax, v);
            }
#pragma unroll
            for (int off = 8; off >= 1; off >>= 1)
                rmax = fmaxf(rmax, __shfl_xor(rmax, off, 32));
            float mnew  = fmaxf(mrow[r], rmax);
            float alpha = __expf(mrow[r] - mnew);
            mrow[r] = mnew;
            float rsum = 0.0f;
#pragma unroll
            for (int ns = 0; ns < 4; ++ns) {
                float p = __expf(s[ns][r] - mnew);
                Ps[wave][(mloc + r) * PS + ns * 16 + nloc] = (_Float16)p;
                rsum += p;
            }
#pragma unroll
            for (int off = 8; off >= 1; off >>= 1)
                rsum += __shfl_xor(rsum, off, 32);
            lrow[r] = alpha * lrow[r] + rsum;
#pragma unroll
            for (int j = 0; j < 4; ++j) oacc[j][r] *= alpha;
        }

        // O += P @ V : A = P (16x32 key chunk), B = V^T rows (contiguous in Vt)
#pragma unroll
        for (int kc = 0; kc < 64; kc += 32) {
            v16h pf = frag_a(&Ps[wave][kc], lane, PS);
#pragma unroll
            for (int n2 = 0; n2 < 4; ++n2) {
                v16h vf = frag_b(&Vt[(n2 * 16) * KS + kc], lane, KS);
                oacc[n2] = __builtin_amdgcn_wmma_f32_16x16x32_f16(
                    false, pf, false, vf, (short)0, oacc[n2], false, false);
            }
        }
    }

    // finalize: O /= l, store as f16 into y[b, t, h*D + d]
#pragma unroll
    for (int r = 0; r < 8; ++r) {
        float inv = 1.0f / lrow[r];
        int q = qRow0 + mloc + r;
        _Float16* dst = Y + (size_t)(b * T + q) * C + h * D;
#pragma unroll
        for (int n2 = 0; n2 < 4; ++n2)
            dst[n2 * 16 + nloc] = (_Float16)(oacc[n2][r] * inv);
    }
}

// ---------------------------------------------------------------------------
extern "C" void kernel_launch(void* const* d_in, const int* in_sizes, int n_in,
                              void* d_out, int out_size, void* d_ws, size_t ws_size,
                              hipStream_t stream) {
    (void)in_sizes; (void)n_in; (void)out_size; (void)ws_size;
    const float* x  = (const float*)d_in[0];   // [B,T,C]
    const float* wa = (const float*)d_in[1];   // [3C,C]
    const float* ba = (const float*)d_in[2];   // [3C]
    const float* wp = (const float*)d_in[3];   // [C,C]
    const float* bp = (const float*)d_in[4];   // [C]
    float* out = (float*)d_out;                // [B,T,C] f32

    constexpr int Bb = 2, T = 2048, C = 1024;
    constexpr size_t nX   = (size_t)Bb * T * C;      // 4,194,304
    constexpr size_t nWA  = (size_t)3 * C * C;       // 3,145,728
    constexpr size_t nWP  = (size_t)C * C;           // 1,048,576
    constexpr size_t nQKV = (size_t)Bb * T * 3 * C;  // 12,582,912

    _Float16* xh   = (_Float16*)d_ws;
    _Float16* wah  = xh + nX;
    _Float16* wph  = wah + nWA;
    _Float16* qkvh = wph + nWP;
    _Float16* yh   = qkvh + nQKV;                    // total ~48 MiB

    cvt_f32_to_f16<<<(int)(nX  / 4 / 256), 256, 0, stream>>>(x,  xh,  (int)(nX  / 4));
    cvt_f32_to_f16<<<(int)(nWA / 4 / 256), 256, 0, stream>>>(wa, wah, (int)(nWA / 4));
    cvt_f32_to_f16<<<(int)(nWP / 4 / 256), 256, 0, stream>>>(wp, wph, (int)(nWP / 4));

    // qkv = x @ w_attn^T + b_attn   (f16 out)
    gemm_bias_wmma<_Float16><<<dim3(3 * C / 128, Bb * T / 128), 256, 0, stream>>>(
        xh, wah, ba, qkvh, Bb * T, 3 * C, C);

    // flash attention -> y (f16)
    flash_attn_wmma<<<Bb * 16 * (T / 128), 256, 0, stream>>>(qkvh, yh);

    // out = y @ w_proj^T + b_proj   (f32 out)
    gemm_bias_wmma<float><<<dim3(C / 128, Bb * T / 128), 256, 0, stream>>>(
        yh, wph, bp, out, Bb * T, C, C);
}
